// MultiScaleSpatialAttention_5136780886063
// MI455X (gfx1250) — matrix operand order repaired and checked by reading, not compiled, not run on hardware
//
#include <hip/hip_runtime.h>
#include <hip/hip_bf16.h>

#define NVOX 400000
#define CCH  64
#define KOFF 27
#define EPSBN 1e-5f
#define NTILES (NVOX / 16)              // 25000
#define WPB 32                          // waves per block (1024 threads)
#define BLOBBYTES ((size_t)KOFF * 2 * 4 * 1024)   // 221184 B per conv

typedef __attribute__((ext_vector_type(16))) __bf16 v16bf;
typedef __attribute__((ext_vector_type(8)))  float  v8f;
typedef unsigned int v4u __attribute__((ext_vector_type(4)));
typedef int v8i __attribute__((ext_vector_type(8)));
typedef int v4i __attribute__((ext_vector_type(4)));

struct alignas(16) U4 { unsigned int x, y, z, w; };
union FragU { U4 u[2]; v16bf v; };

// ---- workspace layout (bytes) ----
static constexpr size_t SZ_PADFEAT = (size_t)(NVOX + 1) * CCH * 2;      // bf16 padded
static constexpr size_t OFF_FPAD   = 0;
static constexpr size_t OFF_XB16   = OFF_FPAD + SZ_PADFEAT;
static constexpr size_t OFF_XF32   = OFF_XB16 + SZ_PADFEAT;
static constexpr size_t OFF_WBLOB  = OFF_XF32 + (size_t)NVOX * CCH * 4;
static constexpr size_t OFF_SCALE  = OFF_WBLOB + 4 * BLOBBYTES;
static constexpr size_t OFF_SHIFT  = OFF_SCALE + 4 * 64 * 4;

// ---- TDM: DMA one conv's swizzled weight blob (221184 B) into LDS ----
// D# group0: count=1, lds_addr, global_addr[56:0], type=2
// D# group1: data_size=8B, 1-row 2D tile of 27648 elements
__device__ __forceinline__ void tdm_load_blob(const void* gsrc, void* ldsdst) {
  unsigned long long ga = (unsigned long long)(size_t)gsrc;
  unsigned lds = (unsigned)(size_t)ldsdst;
  const unsigned n8 = (unsigned)(BLOBBYTES / 8);          // 27648
  v4u g0;
  g0[0] = 1u;                                             // count=1
  g0[1] = lds;                                            // lds_addr
  g0[2] = (unsigned)ga;                                   // global_addr lo
  g0[3] = (unsigned)((ga >> 32) & 0x01FFFFFFu) | (2u << 30); // addr hi | type=2
  v8i g1;
  g1[0] = (int)(3u << 16);                 // workgroup_mask=0, data_size=3 (8B)
  g1[1] = (int)((n8 & 0xFFFFu) << 16);     // tensor_dim0[15:0] @ bits 63:48
  g1[2] = (int)((n8 >> 16) | (1u << 16));  // tensor_dim0[31:16] | tensor_dim1=1
  g1[3] = (int)((n8 & 0xFFFFu) << 16);     // tile_dim0 @ bits 127:112
  g1[4] = 1;                               // tile_dim1=1, tile_dim2=0
  g1[5] = (int)n8;                         // tensor_dim0_stride[31:0]
  g1[6] = 0;
  g1[7] = 0;
  v4i z4 = {0, 0, 0, 0};
  v8i z8 = {0, 0, 0, 0, 0, 0, 0, 0};
  __builtin_amdgcn_tensor_load_to_lds(g0, g1, z4, z4, z8, 0);
}

// ---- prep: feats -> padded bf16, zero pad row of x_init bf16 ----
__global__ void prep_pad(const float* __restrict__ feats,
                         __hip_bfloat16* __restrict__ fpad,
                         __hip_bfloat16* __restrict__ xb16) {
  size_t i = (size_t)blockIdx.x * 256 + threadIdx.x;
  size_t total = (size_t)(NVOX + 1) * CCH;
  if (i >= total) return;
  bool pad = i >= (size_t)NVOX * CCH;
  fpad[i] = __float2bfloat16(pad ? 0.0f : feats[i]);
  if (pad) xb16[i] = __float2bfloat16(0.0f);
}

// ---- prep: swizzle W into CDNA5 16-bit B-matrix (K=32 x N=16) lane layout ----
__global__ void prep_pack_w(const float* __restrict__ Winit,
                            const float* __restrict__ Wbr,
                            __hip_bfloat16* __restrict__ blob) {
  int i = blockIdx.x * 256 + threadIdx.x;
  if (i >= 4 * KOFF * CCH * CCH) return;
  int g = i / (KOFF * CCH * CCH);
  int r = i % (KOFF * CCH * CCH);
  int k = r / (CCH * CCH);
  int cin = (r / CCH) % CCH;
  int cout = r % CCH;
  float w = (g == 0) ? Winit[(size_t)k * CCH * CCH + cin * CCH + cout]
                     : Wbr[(size_t)((g - 1) * KOFF + k) * CCH * CCH + cin * CCH + cout];
  int s = cin >> 5, kk = cin & 31, t = cout >> 4, nn = cout & 15;
  int lane = nn + (((kk >> 3) & 1) << 4);
  int vg = ((kk & 7) + ((kk >> 4) << 3)) >> 1;
  int half = kk & 1;
  size_t dst = ((size_t)(g * KOFF * 2 * 4) + (size_t)(k * 2 + s) * 4 + t) * 512
             + (size_t)lane * 16 + vg * 2 + half;
  blob[dst] = __float2bfloat16(w);
}

// ---- prep: fold conv bias + BN into per-channel scale/shift ----
__global__ void prep_bn(const float* b_i, const float* g_i, const float* be_i,
                        const float* m_i, const float* v_i,
                        const float* b_b, const float* g_b, const float* be_b,
                        const float* m_b, const float* v_b,
                        float* __restrict__ scale, float* __restrict__ shift) {
  int i = threadIdx.x;           // 0..255  (4 convs x 64 ch)
  int g = i >> 6, c = i & 63;
  float gg, bb, be, mm, vv;
  if (g == 0) { gg = g_i[c]; bb = b_i[c]; be = be_i[c]; mm = m_i[c]; vv = v_i[c]; }
  else { int d = g - 1; gg = g_b[d*64+c]; bb = b_b[d*64+c]; be = be_b[d*64+c];
         mm = m_b[d*64+c]; vv = v_b[d*64+c]; }
  float s = gg * rsqrtf(vv + EPSBN);
  scale[i] = s;
  shift[i] = (bb - mm) * s + be;
}

// ---- core: one 16-voxel x 64-ch tile of a 27-tap sparse conv.
// A gathered from global (software-pipelined 1 deep), B read from LDS blob.
__device__ __forceinline__ void conv_tile(const __hip_bfloat16* __restrict__ src_pad,
                                          const int* __restrict__ nbr_k,
                                          const char* smemB,
                                          int tile, int lane, v8f acc[4]) {
  int mrow = lane & 15;
  int hi = lane >> 4;
  const char* lB = smemB + lane * 32;
  int base = tile * 16 + mrow;
  FragU a[2];
  {
    int idx = nbr_k[base];
    const char* rowp = (const char*)src_pad + (size_t)idx * (CCH * 2) + hi * 16;
    a[0].u[0] = *(const U4*)(rowp);
    a[0].u[1] = *(const U4*)(rowp + 32);
    a[1].u[0] = *(const U4*)(rowp + 64);
    a[1].u[1] = *(const U4*)(rowp + 96);
  }
  for (int k = 0; k < KOFF; ++k) {
    FragU an[2];
    if (k + 1 < KOFF) {                       // prefetch next tap's A fragments
      int idx2 = nbr_k[(size_t)(k + 1) * NVOX + base];
      const char* rowp = (const char*)src_pad + (size_t)idx2 * (CCH * 2) + hi * 16;
      an[0].u[0] = *(const U4*)(rowp);
      an[0].u[1] = *(const U4*)(rowp + 32);
      an[1].u[0] = *(const U4*)(rowp + 64);
      an[1].u[1] = *(const U4*)(rowp + 96);
    }
#pragma unroll
    for (int s = 0; s < 2; ++s) {
      const char* bp = lB + (size_t)(k * 2 + s) * 4096;
#pragma unroll
      for (int t = 0; t < 4; ++t) {
        FragU b;
        b.u[0] = *(const U4*)(bp + t * 1024);
        b.u[1] = *(const U4*)(bp + t * 1024 + 16);
        acc[t] = __builtin_amdgcn_wmma_f32_16x16x32_bf16(
            false, a[s].v, false, b.v, (short)0, acc[t], false, false);
      }
    }
    if (k + 1 < KOFF) { a[0] = an[0]; a[1] = an[1]; }
  }
}

// ---- stage 1: conv_init + BN + ReLU -> x_init (f32 + padded bf16) ----
__global__ void __launch_bounds__(1024) stage1_conv_init(
    const __hip_bfloat16* __restrict__ fpad, const int* __restrict__ nbr,
    const __hip_bfloat16* __restrict__ blob,
    const float* __restrict__ scale, const float* __restrict__ shift,
    float* __restrict__ xf32, __hip_bfloat16* __restrict__ xb16) {
  extern __shared__ char smemB[];
  if (threadIdx.x < 32) {                       // wave 0 drives the TDM
    tdm_load_blob(blob, smemB);
    __builtin_amdgcn_s_wait_tensorcnt(0);
  }
  __syncthreads();
  int wave = blockIdx.x * WPB + (threadIdx.x >> 5);
  int lane = threadIdx.x & 31;
  bool valid = wave < NTILES;
  int tile = valid ? wave : NTILES - 1;
  v8f acc[4] = {};
  conv_tile(fpad, nbr, smemB, tile, lane, acc);
  int mbase = tile * 16 + (lane >> 4) * 8;
#pragma unroll
  for (int t = 0; t < 4; ++t) {
    int ch = (lane & 15) + 16 * t;
    float sc = scale[ch], sh = shift[ch];
#pragma unroll
    for (int r = 0; r < 8; ++r) {
      float x = acc[t][r] * sc + sh;
      x = x > 0.f ? x : 0.f;
      if (valid) {
        size_t o = (size_t)(mbase + r) * CCH + ch;
        xf32[o] = x;
        xb16[o] = __float2bfloat16(x);
      }
    }
  }
}

// ---- stage 2: 3 dilated branch convs + BN/ReLU + residual + 1x1 + sigmoid ----
__global__ void __launch_bounds__(1024) stage2_fused(
    const __hip_bfloat16* __restrict__ xb16, const float* __restrict__ xf32,
    const int* __restrict__ nbr, const __hip_bfloat16* __restrict__ blob,
    const float* __restrict__ scale, const float* __restrict__ shift,
    const float* __restrict__ W_out, const float* __restrict__ b_out,
    float* __restrict__ out) {
  extern __shared__ char smemB[];
  int wave = blockIdx.x * WPB + (threadIdx.x >> 5);
  int lane = threadIdx.x & 31;
  bool valid = wave < NTILES;
  int tile = valid ? wave : NTILES - 1;
  int mbase = tile * 16 + (lane >> 4) * 8;
  v8f fused[4];
#pragma unroll
  for (int t = 0; t < 4; ++t) {
    int ch = (lane & 15) + 16 * t;
#pragma unroll
    for (int r = 0; r < 8; ++r)
      fused[t][r] = xf32[(size_t)(mbase + r) * CCH + ch];
  }
  for (int d = 0; d < 3; ++d) {
    __syncthreads();                            // all waves done with prior blob
    if (threadIdx.x < 32) {
      tdm_load_blob((const char*)blob + (size_t)(1 + d) * BLOBBYTES, smemB);
      __builtin_amdgcn_s_wait_tensorcnt(0);
    }
    __syncthreads();                            // blob d staged in LDS
    v8f acc[4] = {};
    conv_tile(xb16, nbr + (size_t)d * KOFF * NVOX, smemB, tile, lane, acc);
    const float* sc_ = scale + (1 + d) * 64;
    const float* sh_ = shift + (1 + d) * 64;
#pragma unroll
    for (int t = 0; t < 4; ++t) {
      int ch = (lane & 15) + 16 * t;
      float sc = sc_[ch], sh = sh_[ch];
#pragma unroll
      for (int r = 0; r < 8; ++r) {
        float x = acc[t][r] * sc + sh;
        fused[t][r] += (x > 0.f ? x : 0.f);
      }
    }
  }
  float w[4];
#pragma unroll
  for (int t = 0; t < 4; ++t) w[t] = W_out[(lane & 15) + 16 * t];
  float bo = b_out[0];
#pragma unroll
  for (int r = 0; r < 8; ++r) {
    float p = fused[0][r] * w[0] + fused[1][r] * w[1]
            + fused[2][r] * w[2] + fused[3][r] * w[3];
    p += __shfl_xor(p, 1, 32);
    p += __shfl_xor(p, 2, 32);
    p += __shfl_xor(p, 4, 32);
    p += __shfl_xor(p, 8, 32);
    if (valid && (lane & 15) == 0) {
      float z = p + bo;
      out[mbase + r] = 1.0f / (1.0f + __expf(-z));
    }
  }
}

extern "C" void kernel_launch(void* const* d_in, const int* in_sizes, int n_in,
                              void* d_out, int out_size, void* d_ws, size_t ws_size,
                              hipStream_t stream) {
  const float* feats  = (const float*)d_in[0];
  const int*   nbr    = (const int*)d_in[1];
  const float* W_init = (const float*)d_in[2];
  const float* b_init = (const float*)d_in[3];
  const float* g_init = (const float*)d_in[4];
  const float* be_init= (const float*)d_in[5];
  const float* m_init = (const float*)d_in[6];
  const float* v_init = (const float*)d_in[7];
  const float* W_br   = (const float*)d_in[8];
  const float* b_br   = (const float*)d_in[9];
  const float* g_br   = (const float*)d_in[10];
  const float* be_br  = (const float*)d_in[11];
  const float* m_br   = (const float*)d_in[12];
  const float* v_br   = (const float*)d_in[13];
  const float* W_out  = (const float*)d_in[14];
  const float* b_out  = (const float*)d_in[15];

  char* ws = (char*)d_ws;
  __hip_bfloat16* fpad = (__hip_bfloat16*)(ws + OFF_FPAD);
  __hip_bfloat16* xb16 = (__hip_bfloat16*)(ws + OFF_XB16);
  float*          xf32 = (float*)(ws + OFF_XF32);
  __hip_bfloat16* blob = (__hip_bfloat16*)(ws + OFF_WBLOB);
  float*          scale= (float*)(ws + OFF_SCALE);
  float*          shift= (float*)(ws + OFF_SHIFT);
  float*          out  = (float*)d_out;

  {
    size_t total = (size_t)(NVOX + 1) * CCH;
    prep_pad<<<(int)((total + 255) / 256), 256, 0, stream>>>(feats, fpad, xb16);
  }
  {
    int total = 4 * KOFF * CCH * CCH;
    prep_pack_w<<<(total + 255) / 256, 256, 0, stream>>>(W_init, W_br, blob);
  }
  prep_bn<<<1, 256, 0, stream>>>(b_init, g_init, be_init, m_init, v_init,
                                 b_br, g_br, be_br, m_br, v_br, scale, shift);

  int blocks = (NTILES + WPB - 1) / WPB;        // 782
  stage1_conv_init<<<blocks, 1024, BLOBBYTES, stream>>>(fpad, nbr, blob, scale,
                                                        shift, xf32, xb16);
  stage2_fused<<<blocks, 1024, BLOBBYTES, stream>>>(xb16, xf32, nbr, blob, scale,
                                                    shift, W_out, b_out, out);
}
